// GraphSAGE_16192026706591
// MI455X (gfx1250) — matrix operand order repaired and checked by reading, not compiled, not run on hardware
//
#include <hip/hip_runtime.h>
#include <math.h>

#define N_NODES  50000
#define N_EDGES  800000
#define HIDDEN   128
#define N_GRAPHS 128
#define N_CLASSES 10

typedef __attribute__((ext_vector_type(16))) __bf16        v16bf;
typedef __attribute__((ext_vector_type(8)))  float         v8f;
typedef __attribute__((ext_vector_type(4)))  unsigned int  u32x4;

union Frag { v16bf v; u32x4 u[2]; };

// ---------------------------------------------------------------------------
// Degree count: cnt[dst[e]] += 1
__global__ void k_count(const int* __restrict__ dst, float* __restrict__ cnt, int E) {
    int e = blockIdx.x * blockDim.x + threadIdx.x;
    if (e < E) atomicAdd(&cnt[dst[e]], 1.0f);
}

// cnt -> 1/max(cnt,1) in place
__global__ void k_inv(float* __restrict__ cnt, int N) {
    int i = blockIdx.x * blockDim.x + threadIdx.x;
    if (i < N) cnt[i] = 1.0f / fmaxf(cnt[i], 1.0f);
}

// Scatter-add: agg[dst[e]][f] += h[src[e]][f]  (thread per edge-feature; a wave
// covers 32 consecutive features of one edge -> uniform indices, 128B gather,
// consecutive-address atomics hitting L2)
__global__ void k_scatter(const float* __restrict__ h, const int* __restrict__ src,
                          const int* __restrict__ dst, float* __restrict__ agg, int total) {
    int i = blockIdx.x * blockDim.x + threadIdx.x;
    if (i >= total) return;
    int e = i >> 7, f = i & 127;
    atomicAdd(&agg[dst[e] * HIDDEN + f], h[src[e] * HIDDEN + f]);
}

// Normalize agg by 1/deg and convert both operands to bf16 for WMMA
__global__ void k_normconv(const float* __restrict__ agg, const float* __restrict__ inv,
                           const float* __restrict__ h,
                           __bf16* __restrict__ abf, __bf16* __restrict__ hbf, int total) {
    int i = blockIdx.x * blockDim.x + threadIdx.x;
    if (i >= total) return;
    int n = i >> 7;
    abf[i] = (__bf16)(agg[i] * inv[n]);
    hbf[i] = (__bf16)h[i];
}

// W[128][128] fp32 row-major -> WT[n][k] bf16 (transposed)
__global__ void k_wprep(const float* __restrict__ W, __bf16* __restrict__ WT) {
    int i = blockIdx.x * blockDim.x + threadIdx.x;
    if (i >= 128 * 128) return;
    int n = i >> 7, k = i & 127;
    WT[i] = (__bf16)W[k * 128 + n];
}

// ---------------------------------------------------------------------------
// Strip WMMA GEMM: out = relu(A1@B1T^T + [A2@B2T^T] + bias)
// A*: [M][128] bf16 row-major.  B*T: [128 out][128 in] bf16 (pre-transposed).
// One wave computes a full 16x128 output strip: A fragment loaded once per
// k-step and reused across all 8 N-tiles (8 f32 accumulators = 64 VGPRs).
// DUAL is compile-time -> single branch-free unrolled body of 64 (or 32) WMMAs.
template <bool DUAL>
__global__ void gemm_strip_bf16_relu(const __bf16* __restrict__ A1,
                                     const __bf16* __restrict__ B1T,
                                     const __bf16* __restrict__ A2,
                                     const __bf16* __restrict__ B2T,
                                     const float*  __restrict__ bias,
                                     float* __restrict__ out,
                                     int M) {
    int wave = threadIdx.x >> 5;
    int lane = threadIdx.x & 31;
    int mt = blockIdx.x * (blockDim.x >> 5) + wave;
    if (mt >= (M >> 4)) return;                 // uniform per wave -> EXEC stays all-1
    int g = lane >> 4, l = lane & 15;

    // A layout (16-bit 16x32): lane group g holds K = g*8..g*8+7 and 16+g*8..+7
    const __bf16* a1p = A1 + (mt * 16 + l) * 128 + g * 8;
    const __bf16* a2p = DUAL ? A2 + (mt * 16 + l) * 128 + g * 8 : a1p;
    // B layout (32x16): lane group g holds K = g*8..g*8+7 and 16+g*8..+7 (same as A)
    const __bf16* b1p = B1T + l * 128 + g * 8;
    const __bf16* b2p = DUAL ? B2T + l * 128 + g * 8 : b1p;

    v8f c[8];
#pragma unroll
    for (int nt = 0; nt < 8; ++nt) c[nt] = (v8f){};

#pragma unroll
    for (int ks = 0; ks < 4; ++ks) {
        Frag a1, a2;
        a1.u[0] = *(const u32x4*)(a1p + ks * 32);
        a1.u[1] = *(const u32x4*)(a1p + ks * 32 + 16);
        if (DUAL) {
            a2.u[0] = *(const u32x4*)(a2p + ks * 32);
            a2.u[1] = *(const u32x4*)(a2p + ks * 32 + 16);
        }
#pragma unroll
        for (int nt = 0; nt < 8; ++nt) {
            Frag b1;
            b1.u[0] = *(const u32x4*)(b1p + nt * 2048 + ks * 32);      // nt*16 rows * 128
            b1.u[1] = *(const u32x4*)(b1p + nt * 2048 + ks * 32 + 16);
            c[nt] = __builtin_amdgcn_wmma_f32_16x16x32_bf16(
                        false, a1.v, false, b1.v, (short)0, c[nt], false, false);
            if (DUAL) {
                Frag b2;
                b2.u[0] = *(const u32x4*)(b2p + nt * 2048 + ks * 32);
                b2.u[1] = *(const u32x4*)(b2p + nt * 2048 + ks * 32 + 16);
                c[nt] = __builtin_amdgcn_wmma_f32_16x16x32_bf16(
                            false, a2.v, false, b2.v, (short)0, c[nt], false, false);
            }
        }
    }

#pragma unroll
    for (int nt = 0; nt < 8; ++nt) {
        float bv = bias[nt * 16 + l];
#pragma unroll
        for (int v = 0; v < 8; ++v) {
            int row = mt * 16 + g * 8 + v;      // C/D layout: VGPR v -> M = v (+8 for lanes 16-31)
            out[row * 128 + nt * 16 + l] = fmaxf(c[nt][v] + bv, 0.0f);
        }
    }
}

// ---------------------------------------------------------------------------
__global__ void k_poolsum(const float* __restrict__ h, const int* __restrict__ batch,
                          float* __restrict__ gsum, int total) {
    int i = blockIdx.x * blockDim.x + threadIdx.x;
    if (i >= total) return;
    int n = i >> 7, f = i & 127;
    atomicAdd(&gsum[batch[n] * HIDDEN + f], h[i]);
}

__global__ void k_gcnt(const int* __restrict__ batch, float* __restrict__ gcnt, int N) {
    int i = blockIdx.x * blockDim.x + threadIdx.x;
    if (i < N) atomicAdd(&gcnt[batch[i]], 1.0f);
}

__global__ void k_gdiv(const float* __restrict__ gsum, const float* __restrict__ gcnt,
                       __bf16* __restrict__ gbf) {
    int i = blockIdx.x * blockDim.x + threadIdx.x;
    if (i >= N_GRAPHS * HIDDEN) return;
    int gph = i >> 7;
    gbf[i] = (__bf16)(gsum[i] / fmaxf(gcnt[gph], 1.0f));
}

// head: logits = ghid @ W2 + b2; log_softmax.  One thread per graph.
__global__ void k_head(const float* __restrict__ ghid, const float* __restrict__ W2,
                       const float* __restrict__ b2, float* __restrict__ out) {
    int t = threadIdx.x;
    if (t >= N_GRAPHS) return;
    const float* hr = ghid + t * HIDDEN;
    float logits[N_CLASSES];
    float mx = -1e30f;
    for (int c = 0; c < N_CLASSES; ++c) {
        float s = b2[c];
        for (int k = 0; k < HIDDEN; ++k) s += hr[k] * W2[k * N_CLASSES + c];
        logits[c] = s;
        mx = fmaxf(mx, s);
    }
    float se = 0.0f;
    for (int c = 0; c < N_CLASSES; ++c) se += expf(logits[c] - mx);
    float lse = logf(se);
    for (int c = 0; c < N_CLASSES; ++c) out[t * N_CLASSES + c] = logits[c] - mx - lse;
}

// ---------------------------------------------------------------------------
extern "C" void kernel_launch(void* const* d_in, const int* in_sizes, int n_in,
                              void* d_out, int out_size, void* d_ws, size_t ws_size,
                              hipStream_t stream) {
    const float* x     = (const float*)d_in[0];
    const int*   eidx  = (const int*)d_in[1];
    const int*   srcI  = eidx;
    const int*   dstI  = eidx + N_EDGES;
    const int*   batch = (const int*)d_in[2];
    const float* Wl[3] = {(const float*)d_in[3], (const float*)d_in[6], (const float*)d_in[9]};
    const float* bl[3] = {(const float*)d_in[4], (const float*)d_in[7], (const float*)d_in[10]};
    const float* Wr[3] = {(const float*)d_in[5], (const float*)d_in[8], (const float*)d_in[11]};
    const float* W1 = (const float*)d_in[12];
    const float* b1 = (const float*)d_in[13];
    const float* W2 = (const float*)d_in[14];
    const float* b2 = (const float*)d_in[15];

    // Workspace carve-up (256B aligned)
    char* base = (char*)d_ws;
    size_t off = 0;
    auto carve = [&](size_t bytes) {
        size_t r = off;
        off = (off + bytes + 255) & ~(size_t)255;
        return r;
    };
    const size_t NH4 = (size_t)N_NODES * HIDDEN * 4;
    const size_t NH2 = (size_t)N_NODES * HIDDEN * 2;
    float*  inv  = (float*)(base + carve((size_t)N_NODES * 4));
    float*  agg  = (float*)(base + carve(NH4));
    float*  hA   = (float*)(base + carve(NH4));
    float*  hB   = (float*)(base + carve(NH4));
    __bf16* abf  = (__bf16*)(base + carve(NH2));
    __bf16* hbf  = (__bf16*)(base + carve(NH2));
    __bf16* wT   = (__bf16*)(base + carve((size_t)7 * 128 * 128 * 2));
    float*  gsum = (float*)(base + carve((size_t)N_GRAPHS * HIDDEN * 4));
    float*  gcnt = (float*)(base + carve((size_t)N_GRAPHS * 4));
    __bf16* gbf  = (__bf16*)(base + carve((size_t)N_GRAPHS * HIDDEN * 2));
    float*  ghid = (float*)(base + carve((size_t)N_GRAPHS * HIDDEN * 4));

    // --- degree normalization (edges fixed across layers: compute once)
    hipMemsetAsync(inv, 0, (size_t)N_NODES * 4, stream);
    k_count<<<(N_EDGES + 255) / 256, 256, 0, stream>>>(dstI, inv, N_EDGES);
    k_inv<<<(N_NODES + 255) / 256, 256, 0, stream>>>(inv, N_NODES);

    // --- weight prep: bf16 transposed copies (order: Wl1,Wr1,Wl2,Wr2,Wl3,Wr3,W1)
    const float* wsrc[7] = {Wl[0], Wr[0], Wl[1], Wr[1], Wl[2], Wr[2], W1};
    for (int m = 0; m < 7; ++m)
        k_wprep<<<64, 256, 0, stream>>>(wsrc[m], wT + (size_t)m * 128 * 128);

    // --- 3 SAGEConv layers
    const float* hcur = x;
    float* houts[3] = {hA, hB, hA};
    const int totNF = N_NODES * HIDDEN;          // 6.4M
    const int totEF = N_EDGES * HIDDEN;          // 102.4M
    const int mtiles = N_NODES / 16;             // 3125 (exact)
    for (int l = 0; l < 3; ++l) {
        hipMemsetAsync(agg, 0, NH4, stream);
        k_scatter<<<(totEF + 255) / 256, 256, 0, stream>>>(hcur, srcI, dstI, agg, totEF);
        k_normconv<<<(totNF + 255) / 256, 256, 0, stream>>>(agg, inv, hcur, abf, hbf, totNF);
        gemm_strip_bf16_relu<true><<<(mtiles + 3) / 4, 128, 0, stream>>>(
            abf, wT + (size_t)(2 * l) * 128 * 128,
            hbf, wT + (size_t)(2 * l + 1) * 128 * 128,
            bl[l], houts[l], N_NODES);
        hcur = houts[l];
    }

    // --- global mean pool
    hipMemsetAsync(gsum, 0, (size_t)N_GRAPHS * HIDDEN * 4, stream);
    hipMemsetAsync(gcnt, 0, (size_t)N_GRAPHS * 4, stream);
    k_poolsum<<<(totNF + 255) / 256, 256, 0, stream>>>(hcur, batch, gsum, totNF);
    k_gcnt<<<(N_NODES + 255) / 256, 256, 0, stream>>>(batch, gcnt, N_NODES);
    k_gdiv<<<(N_GRAPHS * HIDDEN + 255) / 256, 256, 0, stream>>>(gsum, gcnt, gbf);

    // --- MLP layer 1 (WMMA strip, single-source) + head
    gemm_strip_bf16_relu<false><<<((N_GRAPHS / 16) + 3) / 4, 128, 0, stream>>>(
        gbf, wT + (size_t)6 * 128 * 128, (const __bf16*)0, (const __bf16*)0,
        b1, ghid, N_GRAPHS);
    k_head<<<1, 128, 0, stream>>>(ghid, W2, b2, (float*)d_out);
}
